// Attention_20117626815094
// MI455X (gfx1250) — compile-verified, matrix operand order checked by reading, not run on hardware
//
#include <hip/hip_runtime.h>
#include <hip/hip_bf16.h>
#include <stdint.h>

typedef __attribute__((ext_vector_type(16))) __bf16 v16bf;
typedef __attribute__((ext_vector_type(8)))  __bf16 v8bf;
typedef __attribute__((ext_vector_type(8)))  float  v8f;

#define BATCH 8
#define SENC  2048
#define SDEC  2048
#define DDIM  1024

// ---------------- fp32 -> bf16 conversion (8 elems/thread, 128b loads) ----------------
__global__ __launch_bounds__(256) void cvt_bf16_kernel(const float* __restrict__ in,
                                                       __bf16* __restrict__ out,
                                                       long n) {
  long tid = (long)blockIdx.x * blockDim.x + threadIdx.x;
  long stride = (long)gridDim.x * blockDim.x;
  for (long v = tid * 8; v < n; v += stride * 8) {
    float4 a = *(const float4*)&in[v];
    float4 b = *(const float4*)&in[v + 4];
    v8bf o;
    o[0] = (__bf16)a.x; o[1] = (__bf16)a.y; o[2] = (__bf16)a.z; o[3] = (__bf16)a.w;
    o[4] = (__bf16)b.x; o[5] = (__bf16)b.y; o[6] = (__bf16)b.z; o[7] = (__bf16)b.w;
    *(v8bf*)&out[v] = o;
  }
}

// ---------------- tiled fp32 -> bf16 transpose: out[c][r] = in[r][c] ---------------
__global__ __launch_bounds__(256) void transpose_bf16_kernel(const float* __restrict__ in,
                                                             __bf16* __restrict__ out,
                                                             int rows, int cols) {
  __shared__ float tile[32][33];
  const long bo = (long)blockIdx.z * rows * cols;
  const float* src = in + bo;
  __bf16* dst = out + bo;
  const int r0 = blockIdx.y * 32, c0 = blockIdx.x * 32;
  const int tx = threadIdx.x & 31, ty = threadIdx.x >> 5;  // 8 rows per pass
#pragma unroll
  for (int i = ty; i < 32; i += 8) tile[i][tx] = src[(long)(r0 + i) * cols + c0 + tx];
  __syncthreads();
#pragma unroll
  for (int i = ty; i < 32; i += 8)
    dst[(long)(c0 + i) * rows + r0 + tx] = (__bf16)tile[tx][i];
}

// ---------------- async 16B global -> LDS copy (CDNA5, ASYNCcnt-tracked) ----------
__device__ __forceinline__ void async_copy_b128(uint32_t lds_byte_addr, const void* gaddr) {
  asm volatile("global_load_async_to_lds_b128 %0, %1, off"
               :: "v"(lds_byte_addr), "v"(gaddr)
               : "memory");
}
__device__ __forceinline__ void wait_async0() {
  asm volatile("s_wait_asynccnt 0x0" ::: "memory");
}

// stage one 128x32 bf16 tile (row-major, k-contiguous) into LDS at lds_base
__device__ __forceinline__ void stage_tile(const __bf16* __restrict__ src, int ld,
                                           int row0, int k0, uint32_t lds_base, int tid) {
  {
    int c = tid;            // chunk 0..255
    int r = c >> 2, cb = (c & 3) << 4;
    async_copy_b128(lds_base + (uint32_t)(r * 64 + cb),
                    (const char*)(src + (long)(row0 + r) * ld + k0) + cb);
  }
  {
    int c = tid + 256;      // chunk 256..511
    int r = c >> 2, cb = (c & 3) << 4;
    async_copy_b128(lds_base + (uint32_t)(r * 64 + cb),
                    (const char*)(src + (long)(row0 + r) * ld + k0) + cb);
  }
}

// ---------------- bf16 WMMA GEMM: C[M,N] fp32 = A[M,K] * B[K,N] -------------------
// A row-major [M,K] (k-contiguous, lda). B supplied as Bt[n*ldb + k] (k-contiguous).
// Workgroup tile 128x128, 8 wave32 waves, wave tile 64x32 = 4x2 WMMA 16x16x32.
// Double-buffered LDS, filled with global_load_async_to_lds_b128.
__global__ void __launch_bounds__(256)
gemm_bf16_wmma(const __bf16* __restrict__ A, const __bf16* __restrict__ Bt,
               float* __restrict__ C, int K, int lda, int ldb, int ldc) {
  __shared__ __align__(16) __bf16 sm[2][2][128 * 32];   // [buf][A/B][tile] = 32 KB

  const int tid   = threadIdx.x;
  const int lane  = tid & 31;
  const int wave  = tid >> 5;      // 0..7
  const int waveM = wave & 1;      // 2 row groups of 64
  const int waveN = wave >> 1;     // 4 col groups of 32
  const int m0 = blockIdx.y * 128;
  const int n0 = blockIdx.x * 128;

  const int l  = lane & 15;              // M (or N) within 16x16 tile
  const int hk = (lane >> 4) << 3;       // K half-offset: 0 or 8

  const uint32_t smBase = (uint32_t)(uintptr_t)&sm[0][0][0];
  const uint32_t bufStride = 2u * 128u * 32u * 2u;   // bytes per buffer (A+B)
  const uint32_t abStride  = 128u * 32u * 2u;        // bytes per tile

  v8f acc[4][2];
#pragma unroll
  for (int i = 0; i < 4; ++i)
#pragma unroll
    for (int j = 0; j < 2; ++j)
#pragma unroll
      for (int r = 0; r < 8; ++r) acc[i][j][r] = 0.0f;

  const int nIter = K / 32;

  // prologue: stage tile 0 into buffer 0
  stage_tile(A,  lda, m0, 0, smBase, tid);
  stage_tile(Bt, ldb, n0, 0, smBase + abStride, tid);

  for (int it = 0; it < nIter; ++it) {
    const int cur = it & 1;
    wait_async0();        // own async copies (current buffer) complete
    __syncthreads();      // all waves' copies complete & prior reads of other buf done

    if (it + 1 < nIter) {
      const uint32_t nb = smBase + (uint32_t)(1 - cur) * bufStride;
      stage_tile(A,  lda, m0, (it + 1) * 32, nb, tid);
      stage_tile(Bt, ldb, n0, (it + 1) * 32, nb + abStride, tid);
    }

    const __bf16* smA = &sm[cur][0][0];
    const __bf16* smB = &sm[cur][1][0];

    // ---- load fragments (documented 16-bit A 16x32 layout; B mirrored) ----
    v16bf afrag[4], bfrag[2];
#pragma unroll
    for (int i = 0; i < 4; ++i) {
      const __bf16* base = &smA[(waveM * 64 + i * 16 + l) * 32];
      v8bf lo = *(const v8bf*)&base[hk];
      v8bf hi = *(const v8bf*)&base[16 + hk];
#pragma unroll
      for (int j = 0; j < 8; ++j) { afrag[i][j] = lo[j]; afrag[i][8 + j] = hi[j]; }
    }
#pragma unroll
    for (int i = 0; i < 2; ++i) {
      const __bf16* base = &smB[(waveN * 32 + i * 16 + l) * 32];
      v8bf lo = *(const v8bf*)&base[hk];
      v8bf hi = *(const v8bf*)&base[16 + hk];
#pragma unroll
      for (int j = 0; j < 8; ++j) { bfrag[i][j] = lo[j]; bfrag[i][8 + j] = hi[j]; }
    }

    // ---- 4x2 WMMA ----
#pragma unroll
    for (int i = 0; i < 4; ++i)
#pragma unroll
      for (int j = 0; j < 2; ++j)
        acc[i][j] = __builtin_amdgcn_wmma_f32_16x16x32_bf16(
            false, afrag[i], false, bfrag[j], (short)0, acc[i][j], false, false);
  }

  // ---- store C (VGPR r: lanes0-15 -> M=r, lanes16-31 -> M=r+8; N = lane&15) ----
  const int mhalf = (lane >> 4) << 3;
#pragma unroll
  for (int i = 0; i < 4; ++i)
#pragma unroll
    for (int j = 0; j < 2; ++j) {
      int mb = m0 + waveM * 64 + i * 16 + mhalf;
      int nb = n0 + waveN * 32 + j * 16 + l;
#pragma unroll
      for (int r = 0; r < 8; ++r)
        C[(long)(mb + r) * ldc + nb] = acc[i][j][r];
    }
}

// ---------------- row softmax of S[e, 0..SDEC) with transposed bf16 output --------
// PT[t, e] = softmax_t(S[e, :])[t]
__global__ void __launch_bounds__(256)
softmax_rows_T(const float* __restrict__ S, __bf16* __restrict__ PT) {
  const int e = blockIdx.x;
  const float* row = S + (long)e * SDEC;
  __shared__ float red[256];

  float lmax = -3.402823466e+38f;
  for (int t = threadIdx.x; t < SDEC; t += 256) lmax = fmaxf(lmax, row[t]);
  red[threadIdx.x] = lmax;
  __syncthreads();
  for (int s = 128; s > 0; s >>= 1) {
    if (threadIdx.x < s) red[threadIdx.x] = fmaxf(red[threadIdx.x], red[threadIdx.x + s]);
    __syncthreads();
  }
  const float m = red[0];
  __syncthreads();

  float lsum = 0.0f;
  for (int t = threadIdx.x; t < SDEC; t += 256) lsum += __expf(row[t] - m);
  red[threadIdx.x] = lsum;
  __syncthreads();
  for (int s = 128; s > 0; s >>= 1) {
    if (threadIdx.x < s) red[threadIdx.x] += red[threadIdx.x + s];
    __syncthreads();
  }
  const float inv = 1.0f / red[0];

  for (int t = threadIdx.x; t < SDEC; t += 256)
    PT[(long)t * SENC + e] = (__bf16)(__expf(row[t] - m) * inv);
}

extern "C" void kernel_launch(void* const* d_in, const int* in_sizes, int n_in,
                              void* d_out, int out_size, void* d_ws, size_t ws_size,
                              hipStream_t stream) {
  const float* enc = (const float*)d_in[0];   // [B, SENC, D] fp32
  const float* dec = (const float*)d_in[1];   // [B, SDEC, D] fp32
  float* out = (float*)d_out;                 // [B, SDEC, D] fp32

  // workspace layout (120 MB total):
  char* ws = (char*)d_ws;
  __bf16* encb = (__bf16*)ws;                                       // 32 MB [B][e][d]
  __bf16* decb = encb + (size_t)BATCH * SENC * DDIM;                // 32 MB [B][t][d]
  __bf16* encT = decb + (size_t)BATCH * SDEC * DDIM;                // 32 MB [B][d][e]
  float*  S    = (float*)(encT + (size_t)BATCH * SENC * DDIM);      // 16 MB (per batch)
  __bf16* PT   = (__bf16*)(S + (size_t)SENC * SDEC);                //  8 MB (per batch)

  const long nconv = (long)BATCH * SENC * DDIM;
  cvt_bf16_kernel<<<4096, 256, 0, stream>>>(enc, encb, nconv);
  cvt_bf16_kernel<<<4096, 256, 0, stream>>>(dec, decb, nconv);
  transpose_bf16_kernel<<<dim3(DDIM / 32, SENC / 32, BATCH), 256, 0, stream>>>(
      enc, encT, SENC, DDIM);

  for (int b = 0; b < BATCH; ++b) {
    const __bf16* Ae  = encb + (size_t)b * SENC * DDIM;
    const __bf16* Bd  = decb + (size_t)b * SDEC * DDIM;
    const __bf16* BeT = encT + (size_t)b * SENC * DDIM;

    // GEMM1: S[e,t] = sum_d enc[e,d]*dec[t,d]   (M=SENC, N=SDEC, K=DDIM; Bt=dec[t][d])
    gemm_bf16_wmma<<<dim3(SDEC / 128, SENC / 128), 256, 0, stream>>>(
        Ae, Bd, S, DDIM, DDIM, DDIM, SDEC);

    // softmax over t for each e-row; write PT[t,e] bf16
    softmax_rows_T<<<SENC, 256, 0, stream>>>(S, PT);

    // GEMM2: out[t,d] = sum_e PT[t,e]*enc[e,d]  (M=SDEC, N=DDIM, K=SENC; Bt=encT[d][e])
    gemm_bf16_wmma<<<dim3(DDIM / 128, SDEC / 128), 256, 0, stream>>>(
        PT, BeT, out + (size_t)b * SDEC * DDIM, SENC, SENC, SENC, DDIM);
  }
}
